// CombinedAttnProcessor_single_68848325755344
// MI455X (gfx1250) — matrix operand order repaired and checked by reading, not compiled, not run on hardware
//
#include <hip/hip_runtime.h>

// ---------------- problem constants ----------------
#define S_TOK    2304
#define D_MODEL  3072
#define H_HEADS  24
#define HD_DIM   128
#define COND_N   256
#define BLOCK_TOK 2048
#define RANK_R   128
#define T_IP_N   128
#define D_IP_N   4096
#define EPS_F    1e-5f
#define ATTN_SCALE 0.08838834764831845f  // 1/sqrt(128)

typedef __bf16 bf16;
typedef __attribute__((ext_vector_type(16))) __bf16 v16bf;
typedef __attribute__((ext_vector_type(8)))  __bf16 v8bf;
typedef __attribute__((ext_vector_type(8)))  float  v8f;

__device__ inline v8f v8f_zero() {
    v8f z;
#pragma unroll
    for (int i = 0; i < 8; ++i) z[i] = 0.0f;
    return z;
}

// A-fragment (16x32 bf16) from f32 row-major: p -> &A[row*lda + k0 + kh*8]
__device__ inline v16bf load_a_f32(const float* p) {
    v8f f0 = *(const v8f*)p;
    v8f f1 = *(const v8f*)(p + 16);
    v16bf t;
#pragma unroll
    for (int i = 0; i < 8; ++i) { t[i] = (__bf16)f0[i]; t[8 + i] = (__bf16)f1[i]; }
    return t;
}

// B-fragment (32x16 bf16) from f32: p -> &B[n*ldb + k0 + kh*8], two runs of 8 at +0 and +16
__device__ inline v16bf load_b_f32(const float* p) {
    v8f f0 = *(const v8f*)p;
    v8f f1 = *(const v8f*)(p + 16);
    v16bf t;
#pragma unroll
    for (int i = 0; i < 8; ++i) { t[i] = (__bf16)f0[i]; t[8 + i] = (__bf16)f1[i]; }
    return t;
}

__device__ inline v16bf cat8(v8bf lo, v8bf hi) {
    v16bf r;
#pragma unroll
    for (int i = 0; i < 8; ++i) { r[i] = lo[i]; r[8 + i] = hi[i]; }
    return r;
}

// ------------------------------------------------------------------
// GEMM: C[m,n] (+)= sum_k A[m,k] * B[n,k]  (+ bias[n])
// A: f32 (M x K, lda), B: f32 (N x K, ldb) -> on-the-fly bf16, f32 accum.
// Block: 256 thr = 8 waves; tile 128x128; wave tile 32(M) x 64(N).
// grid = (N/128, M/128). All loads of one k-step are batched into
// independent registers so they issue as one clause ahead of the WMMAs.
// ------------------------------------------------------------------
template <bool ACC, bool BIAS>
__global__ __launch_bounds__(256) void gemm_xwT(
    const float* __restrict__ A, int lda,
    const float* __restrict__ B, int ldb,
    float* __restrict__ C, int ldc,
    const float* __restrict__ bias, int K)
{
    const int wave = threadIdx.x >> 5;
    const int lane = threadIdx.x & 31;
    const int wm = wave >> 1;       // 0..3
    const int wn = wave & 1;        // 0..1
    const int m0 = blockIdx.y * 128 + wm * 32;
    const int n0 = blockIdx.x * 128 + wn * 64;
    const int r16 = lane & 15;
    const int kh  = lane >> 4;

    v8f acc[2][4];
#pragma unroll
    for (int i = 0; i < 2; ++i)
#pragma unroll
        for (int j = 0; j < 4; ++j) acc[i][j] = v8f_zero();

    const float* aRow0 = A + (size_t)(m0 + r16) * lda;
    const float* aRow1 = A + (size_t)(m0 + 16 + r16) * lda;
    const float* bRow[4];
#pragma unroll
    for (int ni = 0; ni < 4; ++ni)
        bRow[ni] = B + (size_t)(n0 + ni * 16 + r16) * ldb;

    for (int k0 = 0; k0 < K; k0 += 32) {
        __builtin_prefetch(aRow0 + k0 + 256, 0, 0);  // global_prefetch_b8
        __builtin_prefetch(bRow[0] + k0 + 256, 0, 0);
        // batch all loads of this k-step into independent registers
        v16bf a0 = load_a_f32(aRow0 + k0 + kh * 8);
        v16bf a1 = load_a_f32(aRow1 + k0 + kh * 8);
        v16bf bfr[4];
#pragma unroll
        for (int ni = 0; ni < 4; ++ni)
            bfr[ni] = load_b_f32(bRow[ni] + k0 + kh * 8);
        // then issue the 8 WMMAs
#pragma unroll
        for (int ni = 0; ni < 4; ++ni) {
            acc[0][ni] = __builtin_amdgcn_wmma_f32_16x16x32_bf16(
                false, a0, false, bfr[ni], (short)0, acc[0][ni], false, false);
            acc[1][ni] = __builtin_amdgcn_wmma_f32_16x16x32_bf16(
                false, a1, false, bfr[ni], (short)0, acc[1][ni], false, false);
        }
    }

    const int half = lane >> 4;
#pragma unroll
    for (int mi = 0; mi < 2; ++mi)
#pragma unroll
        for (int ni = 0; ni < 4; ++ni) {
            const int ncol = n0 + ni * 16 + r16;
            const float bval = BIAS ? bias[ncol] : 0.0f;
#pragma unroll
            for (int v = 0; v < 8; ++v) {
                size_t idx = (size_t)(m0 + mi * 16 + half * 8 + v) * ldc + ncol;
                float val = acc[mi][ni][v] + bval;
                if (ACC) C[idx] += val; else C[idx] = val;
            }
        }
}

// ------------------------------------------------------------------
// Per-(token, head) RMSNorm (+opt weight) (+opt RoPE) + cast to bf16.
// One wave per (s, h); lane holds 4 consecutive head-dims.
// ------------------------------------------------------------------
__global__ __launch_bounds__(256) void rmsnorm_rope_cast(
    const float* __restrict__ X, int rows,
    const float* __restrict__ w,
    const float* __restrict__ rc, const float* __restrict__ rs,
    bf16* __restrict__ out)
{
    const int wave = threadIdx.x >> 5;
    const int lane = threadIdx.x & 31;
    const int g = blockIdx.x * 8 + wave;
    const int s = g / H_HEADS;
    const int h = g % H_HEADS;
    if (s >= rows) return;

    const float* xp = X + (size_t)s * D_MODEL + h * HD_DIM + lane * 4;
    float4 x = *(const float4*)xp;
    float ss = x.x * x.x + x.y * x.y + x.z * x.z + x.w * x.w;
#pragma unroll
    for (int off = 16; off >= 1; off >>= 1) ss += __shfl_xor(ss, off, 32);
    const float r = rsqrtf(ss * (1.0f / HD_DIM) + EPS_F);

    float4 y;
    y.x = x.x * r; y.y = x.y * r; y.z = x.z * r; y.w = x.w * r;
    if (w) {
        float4 wv = *(const float4*)(w + lane * 4);
        y.x *= wv.x; y.y *= wv.y; y.z *= wv.z; y.w *= wv.w;
    }
    if (rc) {
        float4 c  = *(const float4*)(rc + (size_t)s * HD_DIM + lane * 4);
        float4 sn = *(const float4*)(rs + (size_t)s * HD_DIM + lane * 4);
        float4 o;
        o.x = y.x * c.x - y.y * sn.x;
        o.y = y.y * c.y + y.x * sn.y;
        o.z = y.z * c.z - y.w * sn.z;
        o.w = y.w * c.w + y.z * sn.w;
        y = o;
    }
    bf16* op = out + (size_t)s * D_MODEL + h * HD_DIM + lane * 4;
    op[0] = (__bf16)y.x; op[1] = (__bf16)y.y; op[2] = (__bf16)y.z; op[3] = (__bf16)y.w;
}

// ------------------------------------------------------------------
// Transpose + cast: in f32 (R x C) -> out bf16 (C x R).
// ------------------------------------------------------------------
__global__ __launch_bounds__(256) void transpose_cast(
    const float* __restrict__ in, int R, int C, bf16* __restrict__ out)
{
    __shared__ float tile[32][33];
    const int c0 = blockIdx.x * 32, r0 = blockIdx.y * 32;
    const int tx = threadIdx.x, ty = threadIdx.y;
#pragma unroll
    for (int k = 0; k < 4; ++k)
        tile[ty + 8 * k][tx] = in[(size_t)(r0 + ty + 8 * k) * C + c0 + tx];
    __syncthreads();
#pragma unroll
    for (int k = 0; k < 4; ++k)
        out[(size_t)(c0 + ty + 8 * k) * R + r0 + tx] = (__bf16)tile[tx][ty + 8 * k];
}

// ------------------------------------------------------------------
// Flash attention. Qb/Kb: bf16 (rows, D_MODEL). VT: bf16 (D_MODEL, nkeys).
// Out: f32 (S, D_MODEL); write or accumulate.
// Mask handled purely by loop bounds: q-rows >= BLOCK use keys
// [condKeyStart, nkeys); others use [0, nkeys).
// grid = (S/128, H); block = 256 (8 waves), wave owns 16 q-rows.
// Pipeline per key tile: batch 8 K-frag loads -> 8 QK WMMAs ->
// issue 8 V-frag loads -> softmax VALU (hides V latency) ->
// P via per-wave LDS (wave-synchronous, no block barrier) -> 8 PV WMMAs.
// ------------------------------------------------------------------
__global__ __launch_bounds__(256) void attn_kernel(
    const bf16* __restrict__ Qb, const bf16* __restrict__ Kb,
    const bf16* __restrict__ VT, float* __restrict__ Out,
    int nkeys, int ldvt, int condKeyStart, float scale, int accumulate)
{
    __shared__ bf16 pbuf[8][16 * 32];
    const int wave = threadIdx.x >> 5;
    const int lane = threadIdx.x & 31;
    const int h  = blockIdx.y;
    const int q0 = blockIdx.x * 128 + wave * 16;
    const int half = lane >> 4;
    const int l16  = lane & 15;

    // Q fragments: 4 x (16x32) covering d = 0..127
    v16bf qf[4];
#pragma unroll
    for (int j = 0; j < 4; ++j) {
        const bf16* qp = Qb + (size_t)(q0 + l16) * D_MODEL + h * HD_DIM + j * 32 + half * 16;
        qf[j] = cat8(*(const v8bf*)qp, *(const v8bf*)(qp + 8));
    }

    v8f o[8];
    float mrun[8], lrun[8];
#pragma unroll
    for (int di = 0; di < 8; ++di) o[di] = v8f_zero();
#pragma unroll
    for (int v = 0; v < 8; ++v) { mrun[v] = -3.0e38f; lrun[v] = 0.0f; }

    const int kstart = (q0 >= BLOCK_TOK) ? condKeyStart : 0;

    for (int kt = kstart; kt < nkeys; kt += 32) {
        // ---- batch all 8 K fragments for this tile ----
        v16bf kfr[2][4];
#pragma unroll
        for (int t = 0; t < 2; ++t)
#pragma unroll
            for (int j = 0; j < 4; ++j)
                kfr[t][j] = *(const v16bf*)(Kb + (size_t)(kt + t * 16 + l16) * D_MODEL
                                            + h * HD_DIM + j * 32 + half * 16);
        // ---- logits: two 16x16 f32 tiles over d = 0..127 ----
        v8f sf[2];
        sf[0] = v8f_zero(); sf[1] = v8f_zero();
#pragma unroll
        for (int t = 0; t < 2; ++t)
#pragma unroll
            for (int j = 0; j < 4; ++j)
                sf[t] = __builtin_amdgcn_wmma_f32_16x16x32_bf16(
                    false, qf[j], false, kfr[t][j], (short)0, sf[t], false, false);

        // ---- issue V fragment loads early; softmax hides their latency ----
        v16bf vfr[8];
#pragma unroll
        for (int di = 0; di < 8; ++di)
            vfr[di] = *(const v16bf*)(VT + (size_t)(h * HD_DIM + di * 16 + l16) * ldvt
                                      + kt + half * 16);

        // ---- online softmax update (per row m = 8*half + v) ----
        float mnew[8], corr[8], ps[8];
#pragma unroll
        for (int v = 0; v < 8; ++v) {
            float s0 = sf[0][v] * scale;
            float s1 = sf[1][v] * scale;
            float rm = fmaxf(s0, s1);
#pragma unroll
            for (int off = 8; off >= 1; off >>= 1) rm = fmaxf(rm, __shfl_xor(rm, off, 32));
            mnew[v] = fmaxf(mrun[v], rm);
            corr[v] = __expf(mrun[v] - mnew[v]);
            float p0 = __expf(s0 - mnew[v]);
            float p1 = __expf(s1 - mnew[v]);
            pbuf[wave][(8 * half + v) * 32 + 0 * 16 + l16] = (__bf16)p0;
            pbuf[wave][(8 * half + v) * 32 + 1 * 16 + l16] = (__bf16)p1;
            float s = p0 + p1;
#pragma unroll
            for (int off = 8; off >= 1; off >>= 1) s += __shfl_xor(s, off, 32);
            ps[v] = s;
        }
#pragma unroll
        for (int v = 0; v < 8; ++v) {
            lrun[v] = lrun[v] * corr[v] + ps[v];
            mrun[v] = mnew[v];
        }
#pragma unroll
        for (int di = 0; di < 8; ++di)
#pragma unroll
            for (int v = 0; v < 8; ++v) o[di][v] *= corr[v];

        // pbuf region is per-wave private; wave32 lanes run in lockstep and
        // same-wave LDS ops complete in order, so only a compiler-level
        // ordering barrier is needed (no block-wide sync).
        __builtin_amdgcn_wave_barrier();
        const bf16* pp = &pbuf[wave][l16 * 32 + half * 16];
        v16bf pf = cat8(*(const v8bf*)pp, *(const v8bf*)(pp + 8));
        __builtin_amdgcn_wave_barrier();

#pragma unroll
        for (int di = 0; di < 8; ++di)
            o[di] = __builtin_amdgcn_wmma_f32_16x16x32_bf16(
                false, pf, false, vfr[di], (short)0, o[di], false, false);
    }

    float inv[8];
#pragma unroll
    for (int v = 0; v < 8; ++v) inv[v] = 1.0f / lrun[v];
#pragma unroll
    for (int di = 0; di < 8; ++di)
#pragma unroll
        for (int v = 0; v < 8; ++v) {
            size_t idx = (size_t)(q0 + 8 * half + v) * D_MODEL + h * HD_DIM + di * 16 + l16;
            float val = o[di][v] * inv[v];
            if (accumulate) Out[idx] += val; else Out[idx] = val;
        }
}

// ------------------------------------------------------------------
extern "C" void kernel_launch(void* const* d_in, const int* in_sizes, int n_in,
                              void* d_out, int out_size, void* d_ws, size_t ws_size,
                              hipStream_t stream)
{
    (void)in_sizes; (void)n_in; (void)out_size; (void)ws_size;
    const float* x    = (const float*)d_in[0];
    const float* img  = (const float*)d_in[1];
    const float* rcos = (const float*)d_in[2];
    const float* rsin = (const float*)d_in[3];
    const float* Wq = (const float*)d_in[4];   const float* bq = (const float*)d_in[5];
    const float* Wk = (const float*)d_in[6];   const float* bk = (const float*)d_in[7];
    const float* Wv = (const float*)d_in[8];   const float* bv = (const float*)d_in[9];
    const float* qd = (const float*)d_in[10];  const float* qu = (const float*)d_in[11];
    const float* kd = (const float*)d_in[12];  const float* ku = (const float*)d_in[13];
    const float* vd = (const float*)d_in[14];  const float* vu = (const float*)d_in[15];
    const float* nqw = (const float*)d_in[16]; const float* nkw = (const float*)d_in[17];
    const float* Wkip = (const float*)d_in[18];
    const float* Wvip = (const float*)d_in[19];
    float* out = (float*)d_out;

    char* p = (char*)d_ws;
    auto alloc = [&](size_t bytes) -> char* {
        char* r = p; p += (bytes + 255) & ~(size_t)255; return r;
    };
    const size_t SD = (size_t)S_TOK * D_MODEL;
    const size_t TD = (size_t)T_IP_N * D_MODEL;
    float* Qf   = (float*)alloc(SD * 4);
    float* Kf   = (float*)alloc(SD * 4);
    float* Vf   = (float*)alloc(SD * 4);
    float* kipf = (float*)alloc(TD * 4);
    float* vipf = (float*)alloc(TD * 4);
    float* tmp  = (float*)alloc((size_t)COND_N * RANK_R * 4);
    bf16* qb    = (bf16*)alloc(SD * 2);
    bf16* kb    = (bf16*)alloc(SD * 2);
    bf16* vT    = (bf16*)alloc(SD * 2);
    bf16* kipb  = (bf16*)alloc(TD * 2);
    bf16* vipT  = (bf16*)alloc(TD * 2);

    const dim3 blk(256);
    const dim3 gMain(D_MODEL / 128, S_TOK / 128);   // 24 x 18
    const dim3 gLoDn(RANK_R / 128, COND_N / 128);   // 1 x 2
    const dim3 gLoUp(D_MODEL / 128, COND_N / 128);  // 24 x 2
    const dim3 gIp(D_MODEL / 128, T_IP_N / 128);    // 24 x 1

    // ---- Q = x @ Wq^T + bq  (+ LoRA on cond rows) ----
    gemm_xwT<false, true ><<<gMain, blk, 0, stream>>>(x, D_MODEL, Wq, D_MODEL, Qf, D_MODEL, bq, D_MODEL);
    gemm_xwT<false, false><<<gLoDn, blk, 0, stream>>>(x + (size_t)BLOCK_TOK * D_MODEL, D_MODEL, qd, D_MODEL, tmp, RANK_R, nullptr, D_MODEL);
    gemm_xwT<true,  false><<<gLoUp, blk, 0, stream>>>(tmp, RANK_R, qu, RANK_R, Qf + (size_t)BLOCK_TOK * D_MODEL, D_MODEL, nullptr, RANK_R);
    // ---- K ----
    gemm_xwT<false, true ><<<gMain, blk, 0, stream>>>(x, D_MODEL, Wk, D_MODEL, Kf, D_MODEL, bk, D_MODEL);
    gemm_xwT<false, false><<<gLoDn, blk, 0, stream>>>(x + (size_t)BLOCK_TOK * D_MODEL, D_MODEL, kd, D_MODEL, tmp, RANK_R, nullptr, D_MODEL);
    gemm_xwT<true,  false><<<gLoUp, blk, 0, stream>>>(tmp, RANK_R, ku, RANK_R, Kf + (size_t)BLOCK_TOK * D_MODEL, D_MODEL, nullptr, RANK_R);
    // ---- V ----
    gemm_xwT<false, true ><<<gMain, blk, 0, stream>>>(x, D_MODEL, Wv, D_MODEL, Vf, D_MODEL, bv, D_MODEL);
    gemm_xwT<false, false><<<gLoDn, blk, 0, stream>>>(x + (size_t)BLOCK_TOK * D_MODEL, D_MODEL, vd, D_MODEL, tmp, RANK_R, nullptr, D_MODEL);
    gemm_xwT<true,  false><<<gLoUp, blk, 0, stream>>>(tmp, RANK_R, vu, RANK_R, Vf + (size_t)BLOCK_TOK * D_MODEL, D_MODEL, nullptr, RANK_R);
    // ---- IP projections: k_ip/v_ip = image_emb @ W^T ----
    gemm_xwT<false, false><<<gIp, blk, 0, stream>>>(img, D_IP_N, Wkip, D_IP_N, kipf, D_MODEL, nullptr, D_IP_N);
    gemm_xwT<false, false><<<gIp, blk, 0, stream>>>(img, D_IP_N, Wvip, D_IP_N, vipf, D_MODEL, nullptr, D_IP_N);

    // ---- RMSNorm (+RoPE for q,k) + bf16 cast ----
    rmsnorm_rope_cast<<<S_TOK * H_HEADS / 8, blk, 0, stream>>>(Qf, S_TOK, nqw, rcos, rsin, qb);
    rmsnorm_rope_cast<<<S_TOK * H_HEADS / 8, blk, 0, stream>>>(Kf, S_TOK, nkw, rcos, rsin, kb);
    rmsnorm_rope_cast<<<T_IP_N * H_HEADS / 8, blk, 0, stream>>>(kipf, T_IP_N, nullptr, nullptr, nullptr, kipb);

    // ---- V transposed to (D, S) bf16 for contiguous PV fragments ----
    transpose_cast<<<dim3(D_MODEL / 32, S_TOK / 32), dim3(32, 8), 0, stream>>>(Vf, S_TOK, D_MODEL, vT);
    transpose_cast<<<dim3(D_MODEL / 32, T_IP_N / 32), dim3(32, 8), 0, stream>>>(vipf, T_IP_N, D_MODEL, vipT);

    // ---- attention: main (block-masked via loop bounds) then IP (accumulate) ----
    attn_kernel<<<dim3(S_TOK / 128, H_HEADS), blk, 0, stream>>>(
        qb, kb, vT, out, S_TOK, S_TOK, BLOCK_TOK, ATTN_SCALE, 0);
    attn_kernel<<<dim3(S_TOK / 128, H_HEADS), blk, 0, stream>>>(
        qb, kipb, vipT, out, T_IP_N, T_IP_N, 0, ATTN_SCALE, 1);
}
